// MultiHeadAttention_78993038508257
// MI455X (gfx1250) — matrix operand order repaired and checked by reading, not compiled, not run on hardware
//
#include <hip/hip_runtime.h>
#include <hip/hip_bf16.h>
#include <stdint.h>

typedef __bf16 bf16;
typedef __attribute__((ext_vector_type(16))) __bf16 v16bf;
typedef __attribute__((ext_vector_type(8)))  __bf16 v8bf;
typedef __attribute__((ext_vector_type(8)))  float   v8f;
typedef __attribute__((ext_vector_type(4)))  unsigned int v4u;
typedef __attribute__((ext_vector_type(8)))  int v8i;
typedef __attribute__((ext_vector_type(4)))  int v4i;

#define DEV __device__ __forceinline__

static constexpr int Bc = 2, Sc = 2048, Dc = 1024, Hc = 16, DKc = 64;
static constexpr int Rc = Bc * Sc;   // 4096 flattened rows

// ---------------------------------------------------------------------------
// WMMA wrapper: D = A(16x32 bf16) * B(32x16 bf16) + C(16x16 f32)
// ---------------------------------------------------------------------------
DEV v8f wmma_bf(v16bf a, v16bf b, v8f c) {
  return __builtin_amdgcn_wmma_f32_16x16x32_bf16(
      /*neg_a=*/false, a, /*neg_b=*/false, b,
      /*c_mod=*/(short)0, c, /*reuse_a=*/false, /*reuse_b=*/false);
}

// A fragment (16x32, MxK) from row-major bf16 matrix, leading dim ld.
// ISA layout: lanes 0-15 -> M=lane, K={k0..k0+7, k0+16..k0+23}
//             lanes16-31 -> M=lane-16, K={k0+8..k0+15, k0+24..k0+31}
DEV v16bf load_a_bf(const bf16* p, int ld, int m0, int k0, int lane) {
  const int hf  = lane >> 4;
  const int row = m0 + (lane & 15);
  const bf16* q = p + row * ld + k0 + hf * 8;
  v8bf lo = *(const v8bf*)(q);
  v8bf hi = *(const v8bf*)(q + 16);
  v16bf a;
#pragma unroll
  for (int i = 0; i < 8; ++i) { a[i] = lo[i]; a[i + 8] = hi[i]; }
  return a;
}

// Same A fragment but source is f32 (convert on the fly, RNE via cast).
DEV v16bf load_a_f32(const float* p, int ld, int m0, int k0, int lane) {
  const int hf  = lane >> 4;
  const int row = m0 + (lane & 15);
  const float* q = p + row * ld + k0 + hf * 8;
  v8f lo = *(const v8f*)(q);
  v8f hi = *(const v8f*)(q + 16);
  v16bf a;
#pragma unroll
  for (int i = 0; i < 8; ++i) { a[i] = (bf16)lo[i]; a[i + 8] = (bf16)hi[i]; }
  return a;
}

// B fragment (32x16, KxN) where B[k][n] = M2[n][k] with M2 row-major, ld.
// ISA layout: lanes 0-15 -> N=lane, K={k0..k0+7, k0+16..k0+23}
//             lanes16-31 -> N=lane-16, K={k0+8..k0+15, k0+24..k0+31}
DEV v16bf load_bt(const bf16* p, int ld, int n0, int k0, int lane) {
  const int n    = n0 + (lane & 15);
  const int koff = k0 + ((lane >> 4) << 3);
  const bf16* q = p + n * ld + koff;
  v8bf lo = *(const v8bf*)(q);
  v8bf hi = *(const v8bf*)(q + 16);
  v16bf b;
#pragma unroll
  for (int i = 0; i < 8; ++i) { b[i] = lo[i]; b[i + 8] = hi[i]; }
  return b;
}

// ---------------------------------------------------------------------------
// Tensor Data Mover: async 2D tile load (tile_x x tile_y of 2-byte elements)
// from global (row stride = stride_elems) into LDS at byte offset lds_off,
// packed contiguously. D# bitfields per CDNA5 ISA 8.3/8.4 (2D: groups 2/3 = 0).
// ---------------------------------------------------------------------------
DEV void tdm_load_2d_bf16(unsigned lds_off, const void* gaddr,
                          unsigned tile_x, unsigned tile_y,
                          unsigned long long stride_elems) {
  const unsigned long long ga = (unsigned long long)(uintptr_t)gaddr;
  const unsigned td0 = 1u << 20, td1 = 1u << 20;  // generous in-bounds dims
  v4u g0;
  g0[0] = 1u;                                     // count=1 (valid descriptor)
  g0[1] = lds_off;                                // lds_addr (bytes)
  g0[2] = (unsigned)(ga & 0xFFFFFFFFu);           // global_addr[31:0]
  g0[3] = (unsigned)((ga >> 32) & 0x01FFFFFFu)    // global_addr[56:32]
          | 0x80000000u;                          // type=2 ("image")
  v8i g1;
  g1[0] = (int)(1u << 16);                                  // data_size=2B
  g1[1] = (int)((td0 & 0xFFFFu) << 16);                     // tensor_dim0 lo
  g1[2] = (int)(((td0 >> 16) & 0xFFFFu) | ((td1 & 0xFFFFu) << 16));
  g1[3] = (int)(((td1 >> 16) & 0xFFFFu) | (tile_x << 16));  // tile_dim0
  g1[4] = (int)(tile_y & 0xFFFFu);                          // tile_dim1
  g1[5] = (int)(stride_elems & 0xFFFFFFFFull);              // dim0_stride lo
  g1[6] = (int)((stride_elems >> 32) & 0xFFFFull);          // dim0_stride hi
  g1[7] = 0;
  v4i z = {0, 0, 0, 0};
#if defined(__clang_major__) && (__clang_major__ >= 23)
  v8i z8 = {0, 0, 0, 0, 0, 0, 0, 0};
  __builtin_amdgcn_tensor_load_to_lds(g0, g1, z, z, z8, 0);
#else
  __builtin_amdgcn_tensor_load_to_lds(g0, g1, z, z, 0);
#endif
}

// ---------------------------------------------------------------------------
// f32 -> bf16 convert (weights)
// ---------------------------------------------------------------------------
__global__ void cvt_bf16_kernel(const float* __restrict__ in,
                                bf16* __restrict__ out, int n) {
  int i = blockIdx.x * blockDim.x + threadIdx.x;
  if (i < n) out[i] = (bf16)in[i];
}

// ---------------------------------------------------------------------------
// GEMM: Y[r, d] = sum_k X[r, k] * W[d, k]
// Each wave computes a 32x64 strip (8 WMMA / k-step); block = 4 waves.
// out_mode 0: bf16 [B,H,S,DK] | 1: bf16 [B,H,DK,S] (V^T) | 2: f32 [R,D]
// ---------------------------------------------------------------------------
template <bool AF32>
__global__ __launch_bounds__(128) void proj_kernel(
    const void* __restrict__ Aptr, const bf16* __restrict__ W,
    void* __restrict__ Out, int out_mode) {
  const int lane = threadIdx.x & 31;
  const int wv   = threadIdx.x >> 5;
  const int m0   = blockIdx.x * 128 + wv * 32;
  const int n0   = blockIdx.y * 64;

  v8f acc[2][4];
#pragma unroll
  for (int m = 0; m < 2; ++m)
#pragma unroll
    for (int t = 0; t < 4; ++t)
#pragma unroll
      for (int r = 0; r < 8; ++r) acc[m][t][r] = 0.0f;

  for (int k0 = 0; k0 < Dc; k0 += 32) {
    v16bf a0, a1;
    if (AF32) {
      const float* X = (const float*)Aptr;
      a0 = load_a_f32(X, Dc, m0, k0, lane);
      a1 = load_a_f32(X, Dc, m0 + 16, k0, lane);
      if (k0 + 32 < Dc)  // hint next A slab into cache (global_prefetch)
        __builtin_prefetch(X + (m0 + (lane & 15)) * Dc + k0 + 32, 0, 3);
    } else {
      const bf16* X = (const bf16*)Aptr;
      a0 = load_a_bf(X, Dc, m0, k0, lane);
      a1 = load_a_bf(X, Dc, m0 + 16, k0, lane);
      if (k0 + 32 < Dc)
        __builtin_prefetch(X + (m0 + (lane & 15)) * Dc + k0 + 32, 0, 3);
    }
#pragma unroll
    for (int t = 0; t < 4; ++t) {
      v16bf b = load_bt(W, Dc, n0 + t * 16, k0, lane);
      acc[0][t] = wmma_bf(a0, b, acc[0][t]);
      acc[1][t] = wmma_bf(a1, b, acc[1][t]);
    }
  }

  const int hf   = lane >> 4;
  const int ncol = lane & 15;
#pragma unroll
  for (int m = 0; m < 2; ++m) {
#pragma unroll
    for (int t = 0; t < 4; ++t) {
      const int d = n0 + t * 16 + ncol;
#pragma unroll
      for (int r = 0; r < 8; ++r) {
        const int row = m0 + m * 16 + r + 8 * hf;
        const float val = acc[m][t][r];
        if (out_mode == 2) {
          ((float*)Out)[(size_t)row * Dc + d] = val;
        } else {
          const int bi = row / Sc, s = row % Sc;
          const int h = d >> 6, dk = d & 63;
          if (out_mode == 0)
            ((bf16*)Out)[(((size_t)(bi * Hc + h)) * Sc + s) * DKc + dk] = (bf16)val;
          else
            ((bf16*)Out)[(((size_t)(bi * Hc + h)) * DKc + dk) * Sc + s] = (bf16)val;
        }
      }
    }
  }
}

// ---------------------------------------------------------------------------
// Causal flash attention per (b,h); block = 4 waves owning 64 query rows.
// K and V^T tiles staged per key-block into LDS by the Tensor Data Mover
// (wave 0 issues tensor_load_to_lds, waits on TENSORcnt, block barriers).
// scores = (Q K^T) * sqrt(DK)  (faithful to reference's inverted scale)
// ---------------------------------------------------------------------------
__global__ __launch_bounds__(128) void attn_kernel(
    const bf16* __restrict__ Qh, const bf16* __restrict__ Kh,
    const bf16* __restrict__ VT, bf16* __restrict__ ctx) {
  __shared__ bf16 sK[64 * 64];      // K tile  [key][dk]
  __shared__ bf16 sV[64 * 64];      // V^T tile [dk][key] (densified by TDM)
  __shared__ bf16 sP[4][16 * 64];   // per-wave P staging

  const int lane = threadIdx.x & 31;
  const int wv   = threadIdx.x >> 5;
  const int mblk = blockIdx.x;   // query block (64 rows)
  const int bh   = blockIdx.y;   // b*H + h

  const bf16* Q = Qh + (size_t)bh * Sc * DKc;
  const bf16* K = Kh + (size_t)bh * Sc * DKc;
  const bf16* V = VT + (size_t)bh * DKc * Sc;

  // LDS byte offsets: flat shared pointers carry the LDS offset in addr[31:0]
  const unsigned sK_off = (unsigned)(uintptr_t)(const void*)&sK[0];
  const unsigned sV_off = (unsigned)(uintptr_t)(const void*)&sV[0];

  const int m0   = mblk * 64 + wv * 16;
  const int hf   = lane >> 4;
  const int ncol = lane & 15;

  // Q A-fragments held in registers for the whole kernel
  const v16bf qa0 = load_a_bf(Q, DKc, m0, 0, lane);
  const v16bf qa1 = load_a_bf(Q, DKc, m0, 32, lane);

  v8f o[4];
#pragma unroll
  for (int t = 0; t < 4; ++t)
#pragma unroll
    for (int r = 0; r < 8; ++r) o[t][r] = 0.0f;

  float mrow[8], lrow[8];
#pragma unroll
  for (int r = 0; r < 8; ++r) { mrow[r] = -3.0e38f; lrow[r] = 0.0f; }

  for (int j = 0; j <= mblk; ++j) {   // causal: skip fully-masked key blocks
    const int kbase = j * 64;

    // ---- TDM: stage K (64x64, stride DK) and V^T (64x64, stride S) ----
    if (wv == 0) {
      tdm_load_2d_bf16(sK_off, K + (size_t)kbase * DKc, 64, 64, DKc);
      tdm_load_2d_bf16(sV_off, V + kbase, 64, 64, Sc);
      __builtin_amdgcn_s_wait_tensorcnt(0);
    }
    __syncthreads();   // tiles visible to all waves

    // ---- scores: 16x64 strip = Q(16x64) * K_j^T (B-frags from LDS) ----
    v8f c[4];
#pragma unroll
    for (int t = 0; t < 4; ++t) {
      v8f cc;
#pragma unroll
      for (int r = 0; r < 8; ++r) cc[r] = 0.0f;
      v16bf b0 = load_bt(sK, DKc, t * 16, 0, lane);
      v16bf b1 = load_bt(sK, DKc, t * 16, 32, lane);
      cc = wmma_bf(qa0, b0, cc);
      cc = wmma_bf(qa1, b1, cc);
      c[t] = cc;
    }

    // ---- scale + causal mask (only the diagonal block is partial) ----
    const bool diag = (j == mblk);
#pragma unroll
    for (int t = 0; t < 4; ++t) {
      const int col = kbase + t * 16 + ncol;
#pragma unroll
      for (int r = 0; r < 8; ++r) {
        float sc = c[t][r] * 8.0f;  // sqrt(DK) = 8
        if (diag && col > (m0 + r + 8 * hf)) sc = -3.0e38f;
        c[t][r] = sc;
      }
    }

    // ---- online softmax: row max over 64 cols (16-lane half reduction) ----
    float alpha[8];
#pragma unroll
    for (int r = 0; r < 8; ++r) {
      float vmax = fmaxf(fmaxf(c[0][r], c[1][r]), fmaxf(c[2][r], c[3][r]));
#pragma unroll
      for (int off = 1; off < 16; off <<= 1)
        vmax = fmaxf(vmax, __shfl_xor(vmax, off, 32));
      const float mnew = fmaxf(mrow[r], vmax);
      alpha[r] = __expf(mrow[r] - mnew);
      mrow[r] = mnew;
    }

    // ---- P = exp(scores - m), accumulate row sums ----
    float rs[8];
#pragma unroll
    for (int r = 0; r < 8; ++r) rs[r] = 0.0f;
#pragma unroll
    for (int t = 0; t < 4; ++t)
#pragma unroll
      for (int r = 0; r < 8; ++r) {
        const float p = __expf(c[t][r] - mrow[r]);
        c[t][r] = p;
        rs[r] += p;
      }
#pragma unroll
    for (int r = 0; r < 8; ++r) {
      float s = rs[r];
#pragma unroll
      for (int off = 1; off < 16; off <<= 1) s += __shfl_xor(s, off, 32);
      lrow[r] = lrow[r] * alpha[r] + s;
    }

    // ---- stage P (bf16) through per-wave LDS; rescale O ----
    bf16* myP = sP[wv];
#pragma unroll
    for (int t = 0; t < 4; ++t) {
#pragma unroll
      for (int r = 0; r < 8; ++r)
        myP[(r + 8 * hf) * 64 + t * 16 + ncol] = (bf16)c[t][r];
#pragma unroll
      for (int r = 0; r < 8; ++r) o[t][r] *= alpha[r];
    }

    // ---- O += P * V_j  (V^T tile already dense in LDS, ld = 64) ----
    v16bf pa0 = load_a_bf(myP, 64, 0, 0, lane);
    v16bf pa1 = load_a_bf(myP, 64, 0, 32, lane);
#pragma unroll
    for (int t = 0; t < 4; ++t) {
      v16bf b0 = load_bt(sV, 64, t * 16, 0, lane);
      v16bf b1 = load_bt(sV, 64, t * 16, 32, lane);
      o[t] = wmma_bf(pa0, b0, o[t]);
      o[t] = wmma_bf(pa1, b1, o[t]);
    }

    __syncthreads();   // all waves done with sK/sV before next TDM overwrite
  }

  // ---- normalize and store ctx (bf16, row-major [B*S, D], col base h*64) ----
  const int bi = bh / Hc, h = bh % Hc;
#pragma unroll
  for (int r = 0; r < 8; ++r) {
    const float inv  = 1.0f / lrow[r];
    const int   srow = m0 + r + 8 * hf;
    bf16* dst = ctx + ((size_t)(bi * Sc + srow)) * Dc + h * DKc;
#pragma unroll
    for (int t = 0; t < 4; ++t) dst[t * 16 + ncol] = (bf16)(o[t][r] * inv);
  }
}

// ---------------------------------------------------------------------------
extern "C" void kernel_launch(void* const* d_in, const int* in_sizes, int n_in,
                              void* d_out, int out_size, void* d_ws, size_t ws_size,
                              hipStream_t stream) {
  const float* q  = (const float*)d_in[0];
  const float* k  = (const float*)d_in[1];
  const float* v  = (const float*)d_in[2];
  // d_in[3] is the mask: it is exactly tril(ones), implemented analytically.
  const float* Wq = (const float*)d_in[4];
  const float* Wk = (const float*)d_in[5];
  const float* Wv = (const float*)d_in[6];
  const float* Wo = (const float*)d_in[7];

  char* ws = (char*)d_ws;
  size_t off = 0;
  auto alloc = [&](size_t bytes) {
    void* p = ws + off;
    off += (bytes + 255) & ~(size_t)255;
    return p;
  };
  bf16* wq_b = (bf16*)alloc((size_t)Dc * Dc * 2);
  bf16* wk_b = (bf16*)alloc((size_t)Dc * Dc * 2);
  bf16* wv_b = (bf16*)alloc((size_t)Dc * Dc * 2);
  bf16* wo_b = (bf16*)alloc((size_t)Dc * Dc * 2);
  bf16* Qh   = (bf16*)alloc((size_t)Rc * Dc * 2);  // [B,H,S,DK]
  bf16* Kh   = (bf16*)alloc((size_t)Rc * Dc * 2);  // [B,H,S,DK]
  bf16* VT   = (bf16*)alloc((size_t)Rc * Dc * 2);  // [B,H,DK,S]
  bf16* ctx  = (bf16*)alloc((size_t)Rc * Dc * 2);  // [B*S, D]

  // weights -> bf16
  const int nw = Dc * Dc;
  const dim3 cb(256), cg((nw + 255) / 256);
  cvt_bf16_kernel<<<cg, cb, 0, stream>>>(Wq, wq_b, nw);
  cvt_bf16_kernel<<<cg, cb, 0, stream>>>(Wk, wk_b, nw);
  cvt_bf16_kernel<<<cg, cb, 0, stream>>>(Wv, wv_b, nw);
  cvt_bf16_kernel<<<cg, cb, 0, stream>>>(Wo, wo_b, nw);

  // projections (A is f32, converted in-register)
  const dim3 pb(128), pg(Rc / 128, Dc / 64);
  proj_kernel<true><<<pg, pb, 0, stream>>>((const void*)q, wq_b, (void*)Qh, 0);
  proj_kernel<true><<<pg, pb, 0, stream>>>((const void*)k, wk_b, (void*)Kh, 0);
  proj_kernel<true><<<pg, pb, 0, stream>>>((const void*)v, wv_b, (void*)VT, 1);

  // causal flash attention (TDM-staged K/V tiles)
  const dim3 ab(128), ag(Sc / 64, Bc * Hc);
  attn_kernel<<<ag, ab, 0, stream>>>(Qh, Kh, VT, ctx);

  // output projection -> f32 d_out
  proj_kernel<false><<<pg, pb, 0, stream>>>((const void*)ctx, wo_b, d_out, 2);
}
